// EnokeeEncoder_10041633538084
// MI455X (gfx1250) — matrix operand order repaired and checked by reading, not compiled, not run on hardware
//
#include <hip/hip_runtime.h>

// ---------------------------------------------------------------------------
// EnokeeEncoder for MI455X (gfx1250, wave32, WMMA + async-to-LDS staging).
// Pipeline: pool -> 4x (QKV gemm, attn, out-proj gemm+resid, LN, FF1 relu gemm,
// FF2 gemm+resid, LN) -> cls1 gemm -> cls2 gemm (N=100000) -> d_out fp32.
// GEMM: bf16 WMMA (v_wmma_f32_16x16x32_bf16), 64x128 block tile, 8 waves of
// 32x32 (4 WMMA/K-step), LDS double-buffered, 1 barrier/step.
// EXACT instantiations (K%32==0, N%128==0) stage tiles with
// global_load_async_to_lds_b128 + s_wait_asynccnt (CDNA5 async path).
// ---------------------------------------------------------------------------

typedef __attribute__((ext_vector_type(16))) __bf16 v16bf;
typedef __attribute__((ext_vector_type(8)))  float  v8f;
typedef __bf16 bf16;

#define NB   32      // batch
#define NS   512     // seq
#define ND   768     // d_model
#define NM   32      // mentions
#define NT   30      // tokens per mention
#define NH   12      // heads
#define NHD  64      // head dim
#define NDFF 3072
#define NL   4
#define NE   100000
#define NROWS (NB*NM)   // 1024

// ------------------------- fp32 -> bf16 conversion -------------------------
__global__ __launch_bounds__(256) void cvt_f32_bf16(const float* __restrict__ src,
                                                    bf16* __restrict__ dst, long n) {
    long i  = (long)blockIdx.x * blockDim.x + threadIdx.x;
    long st = (long)gridDim.x * blockDim.x;
    for (; i < n; i += st) dst[i] = (bf16)src[i];
}

// ------------------------- mention pooling (softmax) -----------------------
__global__ __launch_bounds__(256) void pool_kernel(
    const float* __restrict__ lhs, const float* __restrict__ aw,
    const float* __restrict__ ab, const int* __restrict__ pos,
    const int* __restrict__ emask, float* __restrict__ xf, bf16* __restrict__ xb) {
    int bm = blockIdx.x;               // mention index 0..1023
    int b  = bm >> 5;
    __shared__ float sc[32];
    __shared__ float msk[32];
    int t = threadIdx.x, lane = t & 31, wv = t >> 5;

    for (int tok = wv; tok < NT; tok += 8) {
        bool valid = (emask[bm] != 0);
        for (int j = 0; j <= tok; ++j) valid = valid && (pos[bm * NT + j] != -1);
        float dot = 0.f;
        const float* row = lhs + ((size_t)b * NS + tok) * ND;
        for (int d = lane; d < ND; d += 32) dot += row[d] * aw[d];
        for (int off = 16; off; off >>= 1) dot += __shfl_xor(dot, off, 32);
        if (lane == 0) { msk[tok] = valid ? 1.f : 0.f; sc[tok] = (valid ? dot : 0.f) + ab[0]; }
    }
    __syncthreads();

    float mx = -1e30f;
    for (int i = 0; i < NT; ++i) mx = fmaxf(mx, sc[i]);
    float c[NT]; float sum = 0.f;
    for (int i = 0; i < NT; ++i) { c[i] = __expf(sc[i] - mx); sum += c[i]; }
    float inv = 1.f / sum;
    for (int i = 0; i < NT; ++i) c[i] *= inv * msk[i];

    for (int d = t; d < ND; d += 256) {
        float acc = 0.f;
        for (int i = 0; i < NT; ++i) acc += c[i] * lhs[((size_t)b * NS + i) * ND + d];
        xf[(size_t)bm * ND + d] = acc;
        xb[(size_t)bm * ND + d] = (bf16)acc;
    }
}

// ------------------------- bf16 WMMA GEMM ----------------------------------
// C[Mr,N] = A[Mr,K] * W[N,K]^T (+bias) (+resid) (relu?) -> Cf fp32 / Cb bf16.
union FragBF { v16bf v; uint4 q[2]; };

__device__ __forceinline__ v8f wmma_bf16(const FragBF& a, const FragBF& b, v8f c) {
    return __builtin_amdgcn_wmma_f32_16x16x32_bf16(false, a.v, false, b.v,
                                                   (short)0, c, false, false);
}

// 16B async copy global -> LDS (tracked with ASYNCcnt). lds = LDS byte address.
__device__ __forceinline__ void async_b128(unsigned lds, const bf16* g) {
    unsigned long long ga = (unsigned long long)(size_t)g;
    asm volatile("global_load_async_to_lds_b128 %0, %1, off"
                 :: "v"(lds), "v"(ga) : "memory");
}
__device__ __forceinline__ unsigned lds_addr(const void* p) {
    return (unsigned)(size_t)p;        // generic AS low 32 bits == LDS offset
}

template <bool RELU, bool EXACT>
__global__ __launch_bounds__(256) void gemm_bf16(
    const bf16* __restrict__ A, const bf16* __restrict__ W,
    const float* __restrict__ bias, const float* __restrict__ resid,
    float* __restrict__ Cf, bf16* __restrict__ Cb, int Mr, int N, int K) {
    __shared__ __align__(16) bf16 As[2][64][40];    // +8 pad vs bank conflicts
    __shared__ __align__(16) bf16 Bs[2][128][40];

    const int tid  = threadIdx.x;
    const int lane = tid & 31, wave = tid >> 5;
    const int wm = wave & 1, wn = wave >> 1;          // 2 x 4 wave grid
    const int rowBase = blockIdx.y * 64;
    const int nBase   = blockIdx.x * 128;

    v8f acc00 = {}, acc01 = {}, acc10 = {}, acc11 = {};

    // EXACT staging: 3 async b128 per thread per step, no guards.
    auto stageAsync = [&](int buf, int k0) {
        int r = tid >> 2, c = (tid & 3) * 8;             // A: 64x32
        async_b128(lds_addr(&As[buf][r][c]), A + (size_t)(rowBase + r) * K + k0 + c);
        int r2 = tid >> 1, c2 = (tid & 1) * 16;          // B: 128x32
        const bf16* srcB = W + (size_t)(nBase + r2) * K + k0 + c2;
        async_b128(lds_addr(&Bs[buf][r2][c2]),     srcB);
        async_b128(lds_addr(&Bs[buf][r2][c2 + 8]), srcB + 8);
    };
    // generic staging (classifier shapes), vectorized fast path
    auto stageFull = [&](int buf, int k0) {
        {   int r = tid >> 2, c = (tid & 3) * 8;
            const bf16* src = A + (size_t)(rowBase + r) * K + k0 + c;
            *(uint2*)&As[buf][r][c]     = *(const uint2*)(src);
            *(uint2*)&As[buf][r][c + 4] = *(const uint2*)(src + 4);
        }
        {   int r = tid >> 1, c = (tid & 1) * 16;
            int n = nBase + r;
            if (n < N) {
                const bf16* src = W + (size_t)n * K + k0 + c;
                #pragma unroll
                for (int i = 0; i < 4; ++i)
                    *(uint2*)&Bs[buf][r][c + 4 * i] = *(const uint2*)(src + 4 * i);
            } else {
                const uint2 z = make_uint2(0u, 0u);
                #pragma unroll
                for (int i = 0; i < 4; ++i) *(uint2*)&Bs[buf][r][c + 4 * i] = z;
            }
        }
    };
    // guarded tail staging (K remainder, at most once, outside steady state)
    auto stageTail = [&](int buf, int k0) {
        {   int r = tid >> 2, c = (tid & 3) * 8;
            const bf16* src = A + (size_t)(rowBase + r) * K;
            #pragma unroll
            for (int i = 0; i < 8; ++i)
                As[buf][r][c + i] = (k0 + c + i < K) ? src[k0 + c + i] : (bf16)0.f;
        }
        {   int r = tid >> 1, c = (tid & 1) * 16;
            int n = nBase + r;
            #pragma unroll
            for (int i = 0; i < 16; ++i) {
                bf16 v = (bf16)0.f;
                if (n < N && (k0 + c + i) < K) v = W[(size_t)n * K + k0 + c + i];
                Bs[buf][r][c + i] = v;
            }
        }
    };
    // 4 WMMAs per K-step with fragment reuse
    auto compute = [&](int buf) {
        const int akb = (lane >> 4) << 3;          // A frag K chunk: 0 or 8
        const int bkb = (lane >> 4) << 3;          // B frag K chunk: 0 or 8
        const int ar0 = wm * 32 + (lane & 15), ar1 = ar0 + 16;
        const int br0 = wn * 32 + (lane & 15), br1 = br0 + 16;
        FragBF fa0, fa1, fb0, fb1;
        fa0.q[0] = *(const uint4*)&As[buf][ar0][akb];
        fa0.q[1] = *(const uint4*)&As[buf][ar0][akb + 16];
        fa1.q[0] = *(const uint4*)&As[buf][ar1][akb];
        fa1.q[1] = *(const uint4*)&As[buf][ar1][akb + 16];
        fb0.q[0] = *(const uint4*)&Bs[buf][br0][bkb];
        fb0.q[1] = *(const uint4*)&Bs[buf][br0][bkb + 16];
        fb1.q[0] = *(const uint4*)&Bs[buf][br1][bkb];
        fb1.q[1] = *(const uint4*)&Bs[buf][br1][bkb + 16];
        acc00 = wmma_bf16(fa0, fb0, acc00);
        acc01 = wmma_bf16(fa0, fb1, acc01);
        acc10 = wmma_bf16(fa1, fb0, acc10);
        acc11 = wmma_bf16(fa1, fb1, acc11);
    };

    const int Kmain  = K & ~31;
    const int nsteps = EXACT ? (K >> 5) : ((Kmain >> 5) + ((K > Kmain) ? 1 : 0));

    if constexpr (EXACT) {
        stageAsync(0, 0);
    } else {
        if (Kmain > 0) stageFull(0, 0); else stageTail(0, 0);
    }
    int buf = 0;
    for (int s = 0; s < nsteps; ++s) {
        if constexpr (EXACT)
            asm volatile("s_wait_asynccnt 0x0" ::: "memory");
        __syncthreads();
        if (s + 1 < nsteps) {
            int k0n = (s + 1) << 5;
            if constexpr (EXACT) {
                stageAsync(buf ^ 1, k0n);
            } else {
                if (k0n + 32 <= K) stageFull(buf ^ 1, k0n);
                else               stageTail(buf ^ 1, k0n);
            }
        }
        compute(buf);
        buf ^= 1;
    }

    // epilogue: C/D layout -> N = lane&15, reg r -> M = r + 8*(lane>=16)
    const int nloc = lane & 15;
    const int moff = rowBase + wm * 32 + ((lane >> 4) << 3);
    #pragma unroll
    for (int ni = 0; ni < 2; ++ni) {
        int n = nBase + wn * 32 + ni * 16 + nloc;
        if (!EXACT && n >= N) continue;
        float bv = bias ? bias[n] : 0.f;
        v8f a0 = ni ? acc01 : acc00;   // m-subtile 0
        v8f a1 = ni ? acc11 : acc10;   // m-subtile 1
        #pragma unroll
        for (int r = 0; r < 8; ++r) {
            {
                int m = moff + r;
                float v = a0[r] + bv;
                if (resid) v += resid[(size_t)m * N + n];
                if (RELU) v = fmaxf(v, 0.f);
                if (Cf) Cf[(size_t)m * N + n] = v;
                if (Cb) Cb[(size_t)m * N + n] = (bf16)v;
            }
            {
                int m = moff + 16 + r;
                float v = a1[r] + bv;
                if (resid) v += resid[(size_t)m * N + n];
                if (RELU) v = fmaxf(v, 0.f);
                if (Cf) Cf[(size_t)m * N + n] = v;
                if (Cb) Cb[(size_t)m * N + n] = (bf16)v;
            }
        }
    }
}

// ------------------------- attention (M=32 seq, 1 wave / (b,h)) ------------
__global__ __launch_bounds__(32) void attn_kernel(const bf16* __restrict__ qkv,
                                                  bf16* __restrict__ out) {
    int bh = blockIdx.x; int b = bh / NH, h = bh % NH;
    int m = threadIdx.x;                    // lane == query index
    __shared__ __align__(16) bf16 Ks[32][64];
    __shared__ __align__(16) bf16 Vs[32][64];
    const bf16* base = qkv + ((size_t)b * NM + m) * (3 * ND) + h * NHD;
    const bf16* krow = base + ND;
    const bf16* vrow = base + 2 * ND;
    #pragma unroll
    for (int i = 0; i < 8; ++i) {
        *(uint4*)&Ks[m][i * 8] = *(const uint4*)&krow[i * 8];
        *(uint4*)&Vs[m][i * 8] = *(const uint4*)&vrow[i * 8];
    }
    float q[NHD];
    #pragma unroll
    for (int d = 0; d < NHD; ++d) q[d] = (float)base[d];
    __syncthreads();

    float sc[NM]; float mx = -1e30f;
    for (int k = 0; k < NM; ++k) {
        float s = 0.f;
        #pragma unroll
        for (int d = 0; d < NHD; ++d) s += q[d] * (float)Ks[k][d];
        s *= 0.125f;                       // 1/sqrt(64)
        sc[k] = s; mx = fmaxf(mx, s);
    }
    float sum = 0.f;
    for (int k = 0; k < NM; ++k) { sc[k] = __expf(sc[k] - mx); sum += sc[k]; }
    float inv = 1.f / sum;

    bf16* orow = out + ((size_t)b * NM + m) * ND + h * NHD;
    for (int d = 0; d < NHD; ++d) {
        float o = 0.f;
        for (int k = 0; k < NM; ++k) o += sc[k] * (float)Vs[k][d];
        orow[d] = (bf16)(o * inv);
    }
}

// ------------------------- row LayerNorm (D=768) ---------------------------
__global__ __launch_bounds__(256) void ln_kernel(const float* __restrict__ x,
                                                 const float* __restrict__ g,
                                                 const float* __restrict__ bt,
                                                 float* __restrict__ yf,
                                                 bf16* __restrict__ yb) {
    int row = blockIdx.x;
    const float* xr = x + (size_t)row * ND;
    __shared__ float s1[8], s2[8];
    int t = threadIdx.x, lane = t & 31, wv = t >> 5;
    float a = 0.f, b2 = 0.f;
    for (int d = t; d < ND; d += 256) { float v = xr[d]; a += v; b2 += v * v; }
    for (int off = 16; off; off >>= 1) { a += __shfl_xor(a, off, 32); b2 += __shfl_xor(b2, off, 32); }
    if (lane == 0) { s1[wv] = a; s2[wv] = b2; }
    __syncthreads();
    if (t == 0) {
        float ta = 0.f, tb = 0.f;
        for (int i = 0; i < 8; ++i) { ta += s1[i]; tb += s2[i]; }
        s1[0] = ta; s2[0] = tb;
    }
    __syncthreads();
    float mean = s1[0] * (1.f / ND);
    float var  = s2[0] * (1.f / ND) - mean * mean;
    float inv  = rsqrtf(var + 1e-5f);
    for (int d = t; d < ND; d += 256) {
        float v = (xr[d] - mean) * inv * g[d] + bt[d];
        yf[(size_t)row * ND + d] = v;
        yb[(size_t)row * ND + d] = (bf16)v;
    }
}

// ------------------------- host orchestration ------------------------------
static inline char* wsoff(void* ws, size_t& off, size_t bytes) {
    char* p = (char*)ws + off;
    off = (off + bytes + 255) & ~(size_t)255;
    return p;
}

static void launch_gemm(bool relu, const bf16* A, const bf16* W, const float* bias,
                        const float* resid, float* Cf, bf16* Cb,
                        int Mr, int N, int K, hipStream_t s) {
    dim3 g((N + 127) / 128, Mr / 64), b(256);
    bool exact = ((K & 31) == 0) && ((N & 127) == 0);
    if (exact) {
        if (relu) gemm_bf16<true,  true ><<<g, b, 0, s>>>(A, W, bias, resid, Cf, Cb, Mr, N, K);
        else      gemm_bf16<false, true ><<<g, b, 0, s>>>(A, W, bias, resid, Cf, Cb, Mr, N, K);
    } else {
        if (relu) gemm_bf16<true,  false><<<g, b, 0, s>>>(A, W, bias, resid, Cf, Cb, Mr, N, K);
        else      gemm_bf16<false, false><<<g, b, 0, s>>>(A, W, bias, resid, Cf, Cb, Mr, N, K);
    }
}

static void launch_cvt(const float* src, bf16* dst, long n, hipStream_t s) {
    long blocks = (n + 255) / 256; if (blocks > 8192) blocks = 8192;
    cvt_f32_bf16<<<(int)blocks, 256, 0, s>>>(src, dst, n);
}

extern "C" void kernel_launch(void* const* d_in, const int* in_sizes, int n_in,
                              void* d_out, int out_size, void* d_ws, size_t ws_size,
                              hipStream_t stream) {
    (void)in_sizes; (void)n_in; (void)out_size; (void)ws_size;
    const float* lhs   = (const float*)d_in[0];
    const float* aw    = (const float*)d_in[1];
    const float* ab    = (const float*)d_in[2];
    const float* wi    = (const float*)d_in[3];
    const float* bi    = (const float*)d_in[4];
    const float* wo    = (const float*)d_in[5];
    const float* bo    = (const float*)d_in[6];
    const float* g1    = (const float*)d_in[7];
    const float* b1    = (const float*)d_in[8];
    const float* w1    = (const float*)d_in[9];
    const float* c1    = (const float*)d_in[10];
    const float* w2    = (const float*)d_in[11];
    const float* c2    = (const float*)d_in[12];
    const float* g2    = (const float*)d_in[13];
    const float* b2    = (const float*)d_in[14];
    const float* cw1   = (const float*)d_in[15];
    const float* cw2   = (const float*)d_in[16];
    const float* cb2   = (const float*)d_in[17];
    const int*   pos   = (const int*)d_in[18];
    const int*   emask = (const int*)d_in[19];
    float* out = (float*)d_out;

    // ---- workspace carve-up ----
    size_t off = 0;
    float* xf   = (float*)wsoff(d_ws, off, (size_t)NROWS * ND * 4);
    bf16*  xb   = (bf16*) wsoff(d_ws, off, (size_t)NROWS * ND * 2);
    bf16*  qkvb = (bf16*) wsoff(d_ws, off, (size_t)NROWS * 3 * ND * 2);
    bf16*  attb = (bf16*) wsoff(d_ws, off, (size_t)NROWS * ND * 2);
    float* t0f  = (float*)wsoff(d_ws, off, (size_t)NROWS * ND * 4);
    float* ln1f = (float*)wsoff(d_ws, off, (size_t)NROWS * ND * 4);
    bf16*  ln1b = (bf16*) wsoff(d_ws, off, (size_t)NROWS * ND * 2);
    bf16*  ff1b = (bf16*) wsoff(d_ws, off, (size_t)NROWS * NDFF * 2);
    float* t1f  = (float*)wsoff(d_ws, off, (size_t)NROWS * ND * 4);
    bf16*  clsh = (bf16*) wsoff(d_ws, off, (size_t)NROWS * 100 * 2);
    bf16*  wib  = (bf16*) wsoff(d_ws, off, (size_t)NL * 3 * ND * ND * 2);
    bf16*  wob  = (bf16*) wsoff(d_ws, off, (size_t)NL * ND * ND * 2);
    bf16*  w1b  = (bf16*) wsoff(d_ws, off, (size_t)NL * NDFF * ND * 2);
    bf16*  w2b  = (bf16*) wsoff(d_ws, off, (size_t)NL * ND * NDFF * 2);
    bf16*  c1b  = (bf16*) wsoff(d_ws, off, (size_t)100 * ND * 2);
    bf16*  c2b  = (bf16*) wsoff(d_ws, off, (size_t)NE * 100 * 2);

    // ---- weight conversion fp32 -> bf16 ----
    launch_cvt(wi,  wib, (long)NL * 3 * ND * ND, stream);
    launch_cvt(wo,  wob, (long)NL * ND * ND,     stream);
    launch_cvt(w1,  w1b, (long)NL * NDFF * ND,   stream);
    launch_cvt(w2,  w2b, (long)NL * ND * NDFF,   stream);
    launch_cvt(cw1, c1b, (long)100 * ND,         stream);
    launch_cvt(cw2, c2b, (long)NE * 100,         stream);

    // ---- mention pooling ----
    pool_kernel<<<NROWS, 256, 0, stream>>>(lhs, aw, ab, pos, emask, xf, xb);

    // ---- transformer layers ----
    for (int l = 0; l < NL; ++l) {
        const bf16*  wiL = wib + (size_t)l * 3 * ND * ND;
        const float* biL = bi  + (size_t)l * 3 * ND;
        const bf16*  woL = wob + (size_t)l * ND * ND;
        const float* boL = bo  + (size_t)l * ND;
        const bf16*  w1L = w1b + (size_t)l * NDFF * ND;
        const float* c1L = c1  + (size_t)l * NDFF;
        const bf16*  w2L = w2b + (size_t)l * ND * NDFF;
        const float* c2L = c2  + (size_t)l * ND;

        // QKV projection (+bias) -> bf16
        launch_gemm(false, xb, wiL, biL, nullptr, nullptr, qkvb, NROWS, 3 * ND, ND, stream);
        // attention
        attn_kernel<<<NB * NH, 32, 0, stream>>>(qkvb, attb);
        // out proj + residual(x) -> fp32
        launch_gemm(false, attb, woL, boL, xf, t0f, nullptr, NROWS, ND, ND, stream);
        // LN1
        ln_kernel<<<NROWS, 256, 0, stream>>>(t0f, g1 + (size_t)l * ND, b1 + (size_t)l * ND, ln1f, ln1b);
        // FF1 + relu -> bf16
        launch_gemm(true, ln1b, w1L, c1L, nullptr, nullptr, ff1b, NROWS, NDFF, ND, stream);
        // FF2 + residual(ln1) -> fp32
        launch_gemm(false, ff1b, w2L, c2L, ln1f, t1f, nullptr, NROWS, ND, NDFF, stream);
        // LN2 -> xf/xb (next layer input)
        ln_kernel<<<NROWS, 256, 0, stream>>>(t1f, g2 + (size_t)l * ND, b2 + (size_t)l * ND, xf, xb);
    }

    // ---- classifier: [1024,768]x[100,768]^T -> [1024,100] bf16 ----
    launch_gemm(false, xb, c1b, nullptr, nullptr, nullptr, clsh, NROWS, 100, ND, stream);
    // ---- [1024,100]x[100000,100]^T + bias -> d_out fp32 ----
    launch_gemm(false, clsh, c2b, cb2, nullptr, out, nullptr, NROWS, NE, 100, stream);
}
